// AdvancedWebOfThoughts_71622874628403
// MI455X (gfx1250) — compile-verified, matrix operand order checked by reading, not run on hardware
//
#include <hip/hip_runtime.h>
#include <cstddef>

// ---------------------------------------------------------------------------
// Constants from the reference model
// ---------------------------------------------------------------------------
constexpr int kB  = 8192;   // batch
constexpr int kH  = 256;    // hidden
constexpr int kNN = 8;      // nodes
constexpr int kL  = 4;      // reasoning layers
constexpr int kT  = 32;     // task emb dim
constexpr int kTASK = 2;

typedef __attribute__((ext_vector_type(16))) _Float16 v16h;
typedef __attribute__((ext_vector_type(8)))  _Float16 h8;
typedef __attribute__((ext_vector_type(8)))  float    v8f;
typedef __attribute__((ext_vector_type(4)))  float    f4;

union FragU { v16h v; h8 h[2]; };

__device__ __forceinline__ float gelu_erf(float v) {
    return 0.5f * v * (1.0f + erff(v * 0.70710678118654752f));
}

// ---------------------------------------------------------------------------
// Batched WMMA GEMM:  C[z] = act( A[z % aMod] @ W[z / wDiv]^T + bias[z/wDiv] )
// A: [M,K] row-major, W: [N,K] row-major, C: [M,N] row-major (per batch z).
// Block tile 256x128x32, 256 threads = 8 waves laid out 4(M) x 2(N); each
// wave computes 64x64 as a 4x4 grid of v_wmma_f32_16x16x32_f16 (16 WMMA per
// k-step against 16 ds_load_b128 -> 1.0 LDS-b128 per matrix op).
// f32 -> f16 conversion happens on the LDS staging path.
// Requires M%256==0, N%128==0, K%32==0 (true for every GEMM in this model).
// ---------------------------------------------------------------------------
#define TM 256
#define TN 128
#define TK 32
#define LDK 40   // padded row (halves): keeps 16B alignment, dodges bank conflicts

__global__ __launch_bounds__(256)
void gemm_wmma_kernel(const float* __restrict__ A, int aMod, long aStride,
                      const float* __restrict__ W, int wDiv, long wStride,
                      const float* __restrict__ bias, long bStride,
                      float* __restrict__ C, int M, int N, int K, int act)
{
    __shared__ _Float16 sA[2][TM][LDK];   // 40 KB
    __shared__ _Float16 sB[2][TN][LDK];   // 20 KB

    const int z = blockIdx.z;
    A    += (size_t)(z % aMod) * aStride;
    W    += (size_t)(z / wDiv) * wStride;
    bias += (size_t)(z / wDiv) * bStride;
    C    += (size_t)z * (size_t)M * (size_t)N;

    const int m0   = blockIdx.y * TM;
    const int n0   = blockIdx.x * TN;
    const int tid  = threadIdx.x;
    const int lane = tid & 31;
    const int wave = tid >> 5;
    const int wm   = wave >> 1;   // 0..3 -> 64-row band
    const int wn   = wave & 1;    // 0..1 -> 64-col band

    // staging: thread-pair owns one 16-float half-row; A tile has 256 rows so
    // each pair covers rows lr and lr+128.
    const int lr = tid >> 1;            // 0..127
    const int cb = (tid & 1) << 4;      // 0 or 16

    auto cvt_store = [&](const float* g, _Float16* dst) {
        f4 x0 = *(const f4*)(g);      f4 x1 = *(const f4*)(g + 4);
        f4 x2 = *(const f4*)(g + 8);  f4 x3 = *(const f4*)(g + 12);
        __builtin_prefetch(g + 64, 0, 1);   // two k-tiles ahead hint
        h8 h0, h1;
#pragma unroll
        for (int i = 0; i < 4; ++i) {
            h0[i] = (_Float16)x0[i];  h0[i + 4] = (_Float16)x1[i];
            h1[i] = (_Float16)x2[i];  h1[i + 4] = (_Float16)x3[i];
        }
        *(h8*)(dst)     = h0;
        *(h8*)(dst + 8) = h1;
    };

    auto stage = [&](int buf, int kt) {
        cvt_store(W + (size_t)(n0 + lr) * K + kt * TK + cb, &sB[buf][lr][cb]);
        cvt_store(A + (size_t)(m0 + lr) * K + kt * TK + cb, &sA[buf][lr][cb]);
        cvt_store(A + (size_t)(m0 + lr + 128) * K + kt * TK + cb, &sA[buf][lr + 128][cb]);
    };

    v8f acc[4][4] = {};

    const int nK = K >> 5;
    stage(0, 0);
    __syncthreads();

    // fragment addressing per ISA 16-bit A/B layout: lanes 0-15 row = lane,
    // K chunks at halves {0,16}; lanes 16-31 row = lane-16, chunks at {8,24}.
    const int frow = lane & 15;
    const int fkb  = (lane >> 4) << 3;

    for (int kt = 0; kt < nK; ++kt) {
        const int buf = kt & 1;
        if (kt + 1 < nK) stage(buf ^ 1, kt + 1);

        FragU af[4];
#pragma unroll
        for (int mi = 0; mi < 4; ++mi) {
            af[mi].h[0] = *(const h8*)&sA[buf][wm * 64 + mi * 16 + frow][fkb];
            af[mi].h[1] = *(const h8*)&sA[buf][wm * 64 + mi * 16 + frow][fkb + 16];
        }
#pragma unroll
        for (int ni = 0; ni < 4; ++ni) {
            FragU bf;
            bf.h[0] = *(const h8*)&sB[buf][wn * 64 + ni * 16 + frow][fkb];
            bf.h[1] = *(const h8*)&sB[buf][wn * 64 + ni * 16 + frow][fkb + 16];
#pragma unroll
            for (int mi = 0; mi < 4; ++mi)
                acc[mi][ni] = __builtin_amdgcn_wmma_f32_16x16x32_f16(
                    false, af[mi].v, false, bf.v,
                    (short)0, acc[mi][ni], false, false);
        }
        __syncthreads();
    }

    // epilogue: bias (+ optional GELU).  C/D layout: VGPR r -> row r (+8 for
    // lanes 16-31), col = lane&15 within the 16x16 tile.
    const int roff = (lane >> 4) << 3;
#pragma unroll
    for (int mi = 0; mi < 4; ++mi)
#pragma unroll
        for (int ni = 0; ni < 4; ++ni) {
            const int col = n0 + wn * 64 + ni * 16 + frow;
            const float bv = bias[col];
#pragma unroll
            for (int r = 0; r < 8; ++r) {
                const int row = m0 + wm * 64 + mi * 16 + roff + r;
                float v = acc[mi][ni][r] + bv;
                if (act) v = gelu_erf(v);
                C[(size_t)row * N + col] = v;
            }
        }
}

// ---------------------------------------------------------------------------
// Row LayerNorm (+affine) (+GELU).  One wave per row, wave32 shfl reductions.
// gamma is batched: batch = row / rowsPerBatch, offset batch*gStride.
// ---------------------------------------------------------------------------
__global__ __launch_bounds__(256)
void ln_act_kernel(const float* __restrict__ X, float* __restrict__ Y,
                   const float* __restrict__ gamma, const float* __restrict__ beta,
                   int gStride, long rowsPerBatch, long rows, int cols, int act)
{
    const int lane = threadIdx.x & 31;
    const long row = ((long)blockIdx.x << 3) + (threadIdx.x >> 5);
    if (row >= rows) return;
    const float* x = X + row * cols;
    float*       y = Y + row * cols;
    const int nc = cols >> 5;
    float s = 0.f, s2 = 0.f;
    for (int i = 0; i < nc; ++i) {
        float v = x[lane + (i << 5)];
        s += v; s2 += v * v;
    }
#pragma unroll
    for (int off = 16; off >= 1; off >>= 1) {
        s  += __shfl_xor(s,  off, 32);
        s2 += __shfl_xor(s2, off, 32);
    }
    const float inv = 1.0f / (float)cols;
    const float mu  = s * inv;
    const float var = s2 * inv - mu * mu;
    const float rstd = rsqrtf(var + 1e-5f);
    const long gb = (row / rowsPerBatch) * (long)gStride;
    for (int i = 0; i < nc; ++i) {
        const int c = lane + (i << 5);
        float v = (x[c] - mu) * rstd * gamma[gb + c] + beta[gb + c];
        if (act) v = gelu_erf(v);
        y[c] = v;
    }
}

// ---------------------------------------------------------------------------
// Small-KV multi-head attention: softmax over 8 KV positions, H=256, 4 heads
// of 64.  One wave per (group, batch) row; lane owns 8 contiguous dims of one
// head; per-head dot reductions via 8-lane shfl_xor.
// Q: [G,B,256], K/V: [G,8,B,256], O: [G,B,256]
// ---------------------------------------------------------------------------
__global__ __launch_bounds__(256)
void mha_small_kernel(const float* __restrict__ Q, const float* __restrict__ Kp,
                      const float* __restrict__ Vp, float* __restrict__ O,
                      int nGroups, int B)
{
    const int lane = threadIdx.x & 31;
    const long idx = ((long)blockIdx.x << 3) + (threadIdx.x >> 5);
    if (idx >= (long)nGroups * B) return;
    const int n = (int)(idx / B);
    const int b = (int)(idx % B);
    const int dbase = (lane >> 3) * 64 + (lane & 7) * 8;

    float qv[8];
    const float* q = Q + idx * 256 + dbase;
#pragma unroll
    for (int t = 0; t < 8; ++t) qv[t] = q[t];

    float s[8];
#pragma unroll
    for (int j = 0; j < 8; ++j) {
        const float* k = Kp + ((long)(n * 8 + j) * B + b) * 256 + dbase;
        float p = 0.f;
#pragma unroll
        for (int t = 0; t < 8; ++t) p += qv[t] * k[t];
        p += __shfl_xor(p, 1, 32);
        p += __shfl_xor(p, 2, 32);
        p += __shfl_xor(p, 4, 32);
        s[j] = p * 0.125f;                      // 1/sqrt(64)
    }
    float mx = s[0];
#pragma unroll
    for (int j = 1; j < 8; ++j) mx = fmaxf(mx, s[j]);
    float den = 0.f;
#pragma unroll
    for (int j = 0; j < 8; ++j) { s[j] = expf(s[j] - mx); den += s[j]; }
    const float inv = 1.f / den;

    float o[8] = {};
#pragma unroll
    for (int j = 0; j < 8; ++j) {
        const float* v = Vp + ((long)(n * 8 + j) * B + b) * 256 + dbase;
        const float aw = s[j] * inv;
#pragma unroll
        for (int t = 0; t < 8; ++t) o[t] += aw * v[t];
    }
    float* op = O + idx * 256 + dbase;
#pragma unroll
    for (int t = 0; t < 8; ++t) op[t] = o[t];
}

// ---------------------------------------------------------------------------
// gate = sigmoid(gz . Wg2[n] + bg2[n]);  feats = LN0(feats + gate * o)
// One wave per (node,batch) row of 256.
// ---------------------------------------------------------------------------
__global__ __launch_bounds__(256)
void gate_residual_kernel(float* __restrict__ feats, const float* __restrict__ gz,
                          const float* __restrict__ o, const float* __restrict__ Wg2,
                          const float* __restrict__ bg2, int B)
{
    const int lane = threadIdx.x & 31;
    const long idx = ((long)blockIdx.x << 3) + (threadIdx.x >> 5);
    const int n = (int)(idx / B);
    const float* gzr = gz + idx * 256;
    const float* wg  = Wg2 + n * 256;
    float d = 0.f;
#pragma unroll
    for (int t = 0; t < 8; ++t) { const int c = lane + (t << 5); d += gzr[c] * wg[c]; }
#pragma unroll
    for (int off = 16; off >= 1; off >>= 1) d += __shfl_xor(d, off, 32);
    const float gate = 1.f / (1.f + expf(-(d + bg2[n])));

    float* fr = feats + idx * 256;
    const float* orow = o + idx * 256;
    float xv[8]; float s = 0.f, s2 = 0.f;
#pragma unroll
    for (int t = 0; t < 8; ++t) {
        const int c = lane + (t << 5);
        const float v = fr[c] + gate * orow[c];
        xv[t] = v; s += v; s2 += v * v;
    }
#pragma unroll
    for (int off = 16; off >= 1; off >>= 1) {
        s  += __shfl_xor(s,  off, 32);
        s2 += __shfl_xor(s2, off, 32);
    }
    const float mu  = s * (1.f / 256.f);
    const float var = s2 * (1.f / 256.f) - mu * mu;
    const float rstd = rsqrtf(var + 1e-5f);
#pragma unroll
    for (int t = 0; t < 8; ++t) { const int c = lane + (t << 5); fr[c] = (xv[t] - mu) * rstd; }
}

// ---------------------------------------------------------------------------
// Elementwise helpers
// ---------------------------------------------------------------------------
__global__ void concat_task_kernel(const float* __restrict__ h,
                                   const float* __restrict__ task_emb,
                                   float* __restrict__ xt, long total)
{
    const long i = (long)blockIdx.x * 256 + threadIdx.x;
    if (i >= total) return;
    const long b = i / 288; const int c = (int)(i % 288);
    xt[i] = (c < 256) ? h[b * 256 + c] : task_emb[kTASK * kT + (c - 256)];
}

__global__ void concat_gi_kernel(const float* __restrict__ feats,
                                 const float* __restrict__ o,
                                 float* __restrict__ gi, long total)
{
    const long i = (long)blockIdx.x * 256 + threadIdx.x;
    if (i >= total) return;
    const long r = i / 512; const int c = (int)(i % 512);
    gi[i] = (c < 256) ? feats[r * 256 + c] : o[r * 256 + (c - 256)];
}

__global__ void meanpool_kernel(const float* __restrict__ feats,
                                float* __restrict__ qg, long total)
{
    const long i = (long)blockIdx.x * 256 + threadIdx.x;   // i = b*256 + h
    if (i >= total) return;
    float s = 0.f;
#pragma unroll
    for (int n = 0; n < kNN; ++n) s += feats[(long)n * kB * kH + i];
    qg[i] = s * 0.125f;
}

__global__ void add2_kernel(float* __restrict__ out, const float* __restrict__ a,
                            const float* __restrict__ b, long n)
{
    const long i = (long)blockIdx.x * 256 + threadIdx.x;
    if (i < n) out[i] = a[i] + b[i];
}

__global__ void add3_kernel(float* __restrict__ out, const float* __restrict__ a,
                            const float* __restrict__ b, const float* __restrict__ c,
                            long n)
{
    const long i = (long)blockIdx.x * 256 + threadIdx.x;
    if (i < n) out[i] = a[i] + b[i] + c[i];
}

__global__ __launch_bounds__(256)
void final_dot_kernel(const float* __restrict__ Y, const float* __restrict__ Wf2,
                      const float* __restrict__ bf2, float* __restrict__ out, int B)
{
    const int lane = threadIdx.x & 31;
    const long b = ((long)blockIdx.x << 3) + (threadIdx.x >> 5);
    if (b >= B) return;
    const float* y = Y + b * 128;
    float d = 0.f;
#pragma unroll
    for (int t = 0; t < 4; ++t) { const int c = lane + (t << 5); d += y[c] * Wf2[c]; }
#pragma unroll
    for (int off = 16; off >= 1; off >>= 1) d += __shfl_xor(d, off, 32);
    if (lane == 0) out[b] = d + bf2[0];
}

// ---------------------------------------------------------------------------
// Host-side launch helpers
// ---------------------------------------------------------------------------
static inline void launch_gemm(hipStream_t st, const float* A, int aMod, long aStride,
                               const float* W, int wDiv, long wStride,
                               const float* bias, long bStride, float* C,
                               int M, int N, int K, int batches, int act)
{
    dim3 grid(N / TN, M / TM, batches);
    gemm_wmma_kernel<<<grid, 256, 0, st>>>(A, aMod, aStride, W, wDiv, wStride,
                                           bias, bStride, C, M, N, K, act);
}

static inline void launch_ln(hipStream_t st, const float* X, float* Y,
                             const float* g, const float* b, int gStride,
                             long rowsPerBatch, long rows, int cols, int act)
{
    const unsigned blocks = (unsigned)((rows + 7) / 8);
    ln_act_kernel<<<blocks, 256, 0, st>>>(X, Y, g, b, gStride, rowsPerBatch, rows, cols, act);
}

static inline unsigned ceil256(long n) { return (unsigned)((n + 255) / 256); }

// ---------------------------------------------------------------------------
// Full forward pass
// ---------------------------------------------------------------------------
extern "C" void kernel_launch(void* const* d_in, const int* in_sizes, int n_in,
                              void* d_out, int out_size, void* d_ws, size_t ws_size,
                              hipStream_t stream)
{
    (void)in_sizes; (void)n_in; (void)out_size; (void)ws_size;

    int i = 0;
    const float* x       = (const float*)d_in[i++];   // 0  [B,768]
    const float* W_e1    = (const float*)d_in[i++];   // 1  [512,768]
    const float* b_e1    = (const float*)d_in[i++];
    const float* g_e1    = (const float*)d_in[i++];
    const float* be_e1   = (const float*)d_in[i++];
    const float* W_e2    = (const float*)d_in[i++];   // 5  [256,512]
    const float* b_e2    = (const float*)d_in[i++];
    const float* g_e2    = (const float*)d_in[i++];
    const float* be_e2   = (const float*)d_in[i++];
    const float* task_emb= (const float*)d_in[i++];   // 9  [5,32]
    const float* Wn1     = (const float*)d_in[i++];   // 10 [8,512,288]
    const float* bn1     = (const float*)d_in[i++];
    const float* gn1     = (const float*)d_in[i++];
    const float* bln1    = (const float*)d_in[i++];
    const float* Wn2     = (const float*)d_in[i++];   // 14 [8,256,512]
    const float* bn2     = (const float*)d_in[i++];
    const float* gn2     = (const float*)d_in[i++];
    const float* bln2    = (const float*)d_in[i++];
    const float* Wea_in  = (const float*)d_in[i++];   // 18 [8,768,256]
    const float* bea_in  = (const float*)d_in[i++];   // 19 [8,768]
    const float* Wea_out = (const float*)d_in[i++];   // 20 [8,256,256]
    const float* bea_out = (const float*)d_in[i++];   // 21 [8,256]
    const float* Wg1     = (const float*)d_in[i++];   // 22 [8,256,512]
    const float* bg1     = (const float*)d_in[i++];
    const float* gg1     = (const float*)d_in[i++];
    const float* bgl1    = (const float*)d_in[i++];
    const float* Wg2     = (const float*)d_in[i++];   // 26 [8,1,256]
    const float* bg2     = (const float*)d_in[i++];   // 27 [8]
    const float* Wga_in  = (const float*)d_in[i++];   // 28 [768,256]
    const float* bga_in  = (const float*)d_in[i++];
    const float* Wga_out = (const float*)d_in[i++];   // 30 [256,256]
    const float* bga_out = (const float*)d_in[i++];
    const float* rs_g1   = (const float*)d_in[i++];   // 32 [4,256]
    const float* rs_b1   = (const float*)d_in[i++];
    const float* rs_Win  = (const float*)d_in[i++];   // 34 [4,768,256]
    const float* rs_bin  = (const float*)d_in[i++];   // 35 [4,768]
    const float* rs_Wout = (const float*)d_in[i++];   // 36 [4,256,256]
    const float* rs_bout = (const float*)d_in[i++];
    const float* rs_g2   = (const float*)d_in[i++];
    const float* rs_b2   = (const float*)d_in[i++];
    const float* rs_Wf1  = (const float*)d_in[i++];   // 40 [4,1024,256]
    const float* rs_bf1  = (const float*)d_in[i++];
    const float* rs_Wf2  = (const float*)d_in[i++];   // 42 [4,256,1024]
    const float* rs_bf2  = (const float*)d_in[i++];
    const float* Wo1     = (const float*)d_in[i++];   // 44 [256,256]
    const float* bo1     = (const float*)d_in[i++];
    const float* go1     = (const float*)d_in[i++];
    const float* bol1    = (const float*)d_in[i++];
    const float* Wo2     = (const float*)d_in[i++];   // 48 [128,256]
    const float* bo2     = (const float*)d_in[i++];
    const float* Wf1     = (const float*)d_in[i++];   // 50 [128,128]
    const float* bf1     = (const float*)d_in[i++];
    const float* gf1     = (const float*)d_in[i++];
    const float* bfl1    = (const float*)d_in[i++];
    const float* Wf2     = (const float*)d_in[i++];   // 54 [1,128]
    const float* bf2     = (const float*)d_in[i++];   // 55 [1]

    // ---------------- workspace layout (floats), with deliberate aliasing ---
    float* ws = (float*)d_ws;
    size_t off = 0;
    auto alloc = [&](size_t n) { float* p = ws + off; off += n; return p; };

    float* t2h  = alloc((size_t)kB * 512);            // embedding hidden
    float* hb   = alloc((size_t)kB * 256);
    float* xt   = alloc((size_t)kB * 288);
    float* feats= alloc((size_t)kNN * kB * kH);
    float* bufZ = alloc((size_t)kNN * kB * 512);      // z / gi ; aliases K,V
    float* bufO = alloc((size_t)kNN * kB * kH);       // out-proj results
    float* bufA = alloc((size_t)kNN * kB * kH);       // raw attn ; aliases gz
    float* qn   = alloc((size_t)kB * kH);
    float* stA  = alloc((size_t)kB * kH);
    float* s1b  = alloc((size_t)kB * kH);
    float* t1   = alloc((size_t)kB * kH);
    float* t2   = alloc((size_t)kB * kH);
    float* ffb  = alloc((size_t)kB * 1024);
    float* y1   = alloc((size_t)kB * 256);
    float* y2   = alloc((size_t)kB * 128);
    float* y3   = alloc((size_t)kB * 128);
    float* go   = alloc((size_t)kB * kH);

    float* kb = bufZ;                                 // [8,B,H]  (round-local)
    float* vb = bufZ + (size_t)kNN * kB * kH;         // [8,B,H]
    float* gz = bufA;                                 // reuse after out-proj

    // ---------------- 1) embedding MLP -------------------------------------
    launch_gemm(stream, x, 1, 0, W_e1, 1, 0, b_e1, 0, t2h, kB, 512, 768, 1, 0);
    launch_ln(stream, t2h, t2h, g_e1, be_e1, 0, kB, kB, 512, 1);
    launch_gemm(stream, t2h, 1, 0, W_e2, 1, 0, b_e2, 0, hb, kB, 256, 512, 1, 0);
    launch_ln(stream, hb, hb, g_e2, be_e2, 0, kB, kB, 256, 1);
    concat_task_kernel<<<ceil256((long)kB * 288), 256, 0, stream>>>(hb, task_emb, xt, (long)kB * 288);

    // ---------------- 2) per-node transforms (batched over grid.z) ---------
    launch_gemm(stream, xt, 1, 0, Wn1, 1, 512L * 288, bn1, 512, bufZ, kB, 512, 288, kNN, 0);
    launch_ln(stream, bufZ, bufZ, gn1, bln1, 512, kB, (long)kNN * kB, 512, 1);
    launch_gemm(stream, bufZ, kNN, (long)kB * 512, Wn2, 1, 256L * 512, bn2, 256, feats, kB, 256, 512, kNN, 0);
    launch_ln(stream, feats, feats, gn2, bln2, 256, kB, (long)kNN * kB, 256, 1);

    // ---------------- 3) three message-passing rounds -----------------------
    for (int r = 0; r < 3; ++r) {
        for (int n = 0; n < kNN; ++n) {
            const float* Wq = Wea_in + (size_t)n * 768 * 256;
            const float* Wk = Wq + 256 * 256;
            const float* Wv = Wq + 2 * 256 * 256;
            const float* bq = bea_in + (size_t)n * 768;
            // Q_n = feats[n] @ Wq^T
            launch_gemm(stream, feats + (size_t)n * kB * kH, 1, 0, Wq, 1, 0, bq, 0,
                        qn, kB, 256, 256, 1, 0);
            // K_n[j] = feats[j] @ Wk^T  (batched over j)
            launch_gemm(stream, feats, kNN, (long)kB * kH, Wk, kNN, 0, bq + 256, 0,
                        kb, kB, 256, 256, kNN, 0);
            // V_n[j] = feats[j] @ Wv^T
            launch_gemm(stream, feats, kNN, (long)kB * kH, Wv, kNN, 0, bq + 512, 0,
                        vb, kB, 256, 256, kNN, 0);
            // softmax-attention over the 8 node positions
            mha_small_kernel<<<kB / 8, 256, 0, stream>>>(qn, kb, vb,
                        bufA + (size_t)n * kB * kH, 1, kB);
        }
        // out-projection, batched over nodes
        launch_gemm(stream, bufA, kNN, (long)kB * kH, Wea_out, 1, 256L * 256,
                    bea_out, 256, bufO, kB, 256, 256, kNN, 0);
        // gi = [feats, o]
        concat_gi_kernel<<<ceil256((long)kNN * kB * 512), 256, 0, stream>>>(
            feats, bufO, bufZ, (long)kNN * kB * 512);
        // gz = GELU(LN(gi @ Wg1^T + bg1))
        launch_gemm(stream, bufZ, kNN, (long)kB * 512, Wg1, 1, 256L * 512,
                    bg1, 256, gz, kB, 256, 512, kNN, 0);
        launch_ln(stream, gz, gz, gg1, bgl1, 256, kB, (long)kNN * kB, 256, 1);
        // gated residual + LN0
        gate_residual_kernel<<<(kNN * kB) / 8, 256, 0, stream>>>(feats, gz, bufO, Wg2, bg2, kB);
    }

    // ---------------- 4) global attention -----------------------------------
    meanpool_kernel<<<ceil256((long)kB * kH), 256, 0, stream>>>(feats, t1, (long)kB * kH);
    launch_gemm(stream, t1, 1, 0, Wga_in, 1, 0, bga_in, 0, qn, kB, 256, 256, 1, 0);
    launch_gemm(stream, feats, kNN, (long)kB * kH, Wga_in + 256 * 256, kNN, 0,
                bga_in + 256, 0, kb, kB, 256, 256, kNN, 0);
    launch_gemm(stream, feats, kNN, (long)kB * kH, Wga_in + 512 * 256, kNN, 0,
                bga_in + 512, 0, vb, kB, 256, 256, kNN, 0);
    mha_small_kernel<<<kB / 8, 256, 0, stream>>>(qn, kb, vb, go, 1, kB);
    launch_gemm(stream, go, 1, 0, Wga_out, 1, 0, bga_out, 0, stA, kB, 256, 256, 1, 0);

    // ---------------- 5) reasoning layers (seq_len==1 => attn = out(v)) -----
    for (int l = 0; l < kL; ++l) {
        launch_ln(stream, stA, t1, rs_g1 + l * 256, rs_b1 + l * 256, 0, kB, kB, 256, 0);
        launch_gemm(stream, t1, 1, 0, rs_Win + (size_t)l * 768 * 256 + 512 * 256, 1, 0,
                    rs_bin + l * 768 + 512, 0, t2, kB, 256, 256, 1, 0);
        launch_gemm(stream, t2, 1, 0, rs_Wout + (size_t)l * 256 * 256, 1, 0,
                    rs_bout + l * 256, 0, t1, kB, 256, 256, 1, 0);
        add2_kernel<<<ceil256((long)kB * 256), 256, 0, stream>>>(s1b, stA, t1, (long)kB * 256);
        launch_ln(stream, s1b, t1, rs_g2 + l * 256, rs_b2 + l * 256, 0, kB, kB, 256, 0);
        launch_gemm(stream, t1, 1, 0, rs_Wf1 + (size_t)l * 1024 * 256, 1, 0,
                    rs_bf1 + l * 1024, 0, ffb, kB, 1024, 256, 1, 1 /*gelu*/);
        launch_gemm(stream, ffb, 1, 0, rs_Wf2 + (size_t)l * 256 * 1024, 1, 0,
                    rs_bf2 + l * 256, 0, t2, kB, 256, 1024, 1, 0);
        add3_kernel<<<ceil256((long)kB * 256), 256, 0, stream>>>(stA, stA, s1b, t2, (long)kB * 256);
    }

    // ---------------- 6) output head + final numeric -------------------------
    launch_gemm(stream, stA, 1, 0, Wo1, 1, 0, bo1, 0, y1, kB, 256, 256, 1, 0);
    launch_ln(stream, y1, y1, go1, bol1, 0, kB, kB, 256, 1);
    launch_gemm(stream, y1, 1, 0, Wo2, 1, 0, bo2, 0, y2, kB, 128, 256, 1, 0);
    launch_gemm(stream, y2, 1, 0, Wf1, 1, 0, bf1, 0, y3, kB, 128, 128, 1, 0);
    launch_ln(stream, y3, y3, gf1, bfl1, 0, kB, kB, 128, 1);
    final_dot_kernel<<<kB / 8, 256, 0, stream>>>(y3, Wf2, bf2, (float*)d_out, kB);
}
